// Transformer_67095979098405
// MI455X (gfx1250) — matrix operand order repaired and checked by reading, not compiled, not run on hardware
//
#include <hip/hip_runtime.h>
#include <math.h>

// ---------------------------------------------------------------------------
// MI455X (gfx1250) transformer forward.
//   - bf16 WMMA (v_wmma_f32_16x16x32_bf16) for every matmul, f32 accumulate.
//   - Weights (fp32) streamed through LDS once per call (workgroup covers the
//     whole M=1024 token dimension -> each weight element fetched once).
//   - Activations bf16 (1.5 MB) -> resident in 192 MB L2.
//   - GEMM templated on flags (straight-line specializations), software
//     pipelined: ping-pong A-fragment arrays + deferred-pack B loads, LDS
//     B tile double buffered (one barrier per 32-wide K chunk).
//   - Causal attention skips fully-masked score tiles and zero-prob K range.
// ---------------------------------------------------------------------------

typedef unsigned short u16;
typedef __bf16 bf16_t;
typedef bf16_t v16bf __attribute__((ext_vector_type(16)));
typedef float  v8f   __attribute__((ext_vector_type(8)));

union FragBF { v16bf v; uint4 q[2]; u16 u[16]; };

__device__ __forceinline__ u16 f2bf(float f) {
  unsigned int u = __float_as_uint(f);
  return (u16)((u + 0x7FFFu + ((u >> 16) & 1u)) >> 16);   // round-to-nearest-even
}

#define GF_F32    1    // store f32 C
#define GF_BF16   2    // store bf16 C
#define GF_TRANS  4    // store bf16 C transposed (Ct[n*ldct + m])
#define GF_RELU   8
#define GF_BIAS   16
#define GF_BBF16  32   // B operand is bf16 (activations) instead of f32 (weights)
#define GF_CTRIU  64   // causal scores: wave skips compute when n0 > m0+127
#define GF_CKLIM  128  // causal P@V: limit K loop to m0+128 (probs beyond are 0)

// ---------------------------------------------------------------------------
// Generic GEMM:  C[m,n] = sum_k A[m,k] * B[n,k]  (i.e. A @ B^T)
//   A: bf16, row-major, leading dim lda, per-z element offset aOffZ
//   B: fp32 weights or bf16 activations, row-major [N,K], ldb, per-z bOffZ
//   M fixed at 1024: block = 8 waves, wave w covers rows w*128..w*128+127
//   (8 vertical 16x16 tiles); B tile hits LDS once per workgroup.
// grid = (N/16, 1, Z), block = (32, 8)
// ---------------------------------------------------------------------------
template <int FLAGS>
__global__ __launch_bounds__(256)
void wmma_gemm(const u16* __restrict__ A, int lda, long aOffZ,
               const void* __restrict__ Bp, int ldb, long bOffZ,
               const float* __restrict__ bias,
               float* __restrict__ Cf, u16* __restrict__ Cbf, u16* __restrict__ Ct,
               int ldc, long cOffZ, int ldct,
               int K)
{
  __shared__ __align__(16) u16 Bs[2][16 * 32];  // double-buffered 16x32 B tile

  const int lane = threadIdx.x;        // 0..31
  const int wave = threadIdx.y;        // 0..7
  const int tid  = wave * 32 + lane;   // 0..255
  const int n0   = blockIdx.x * 16;
  const int z    = blockIdx.z;
  const int half = lane >> 4;          // 0 or 1
  const int l15  = lane & 15;
  const int m0   = wave * 128;         // 8 tiles of 16 rows per wave (M==1024)

  const u16* Abase = A + (long)z * aOffZ;

  // cooperative-load source coordinates: 2 consecutive elements per thread
  const int e0 = tid * 2;              // 0..510 (512 bf16 elements per tile)
  const int br = e0 >> 5;              // B row within tile (0..15)
  const int bc = e0 & 31;              // k within chunk (even)

  const u16*   Bb16 = (const u16*)Bp   + (long)z * bOffZ + (long)(n0 + br) * ldb + bc;
  const float* Bf32 = (const float*)Bp + (long)z * bOffZ + (long)(n0 + br) * ldb + bc;

  // causal controls (wave-uniform)
  const bool wactive = (FLAGS & GF_CTRIU) ? (n0 <= m0 + 127) : true;
  const int  Keff    = (FLAGS & GF_CKLIM) ? ((m0 + 128 < K) ? m0 + 128 : K) : K;

  v8f acc[8];
#pragma unroll
  for (int t = 0; t < 8; ++t)
#pragma unroll
    for (int j = 0; j < 8; ++j) acc[t][j] = 0.0f;

  // ---- pipelined B: issue raw load now, pack to bf16 one iteration later --
  unsigned int braw16; float2 brawf;
  auto issueB = [&](int k) {
    if (FLAGS & GF_BBF16) braw16 = *(const unsigned int*)(Bb16 + k);
    else                  brawf  = *(const float2*)(Bf32 + k);
  };
  auto packB = [&]() -> unsigned int {
    if (FLAGS & GF_BBF16) return braw16;
    return (unsigned int)f2bf(brawf.x) | ((unsigned int)f2bf(brawf.y) << 16);
  };

  // ---- pipelined A: ping-pong fragment arrays (no per-iteration copies) ---
  FragBF afA[8], afB[8];
  auto issueA = [&](FragBF (&dst)[8], int k) {
#pragma unroll
    for (int t = 0; t < 8; ++t) {
      const u16* ar = Abase + (long)(m0 + 16 * t + l15) * lda + k + half * 8;
      dst[t].q[0] = *(const uint4*)ar;         // K {0..7} / {8..15}
      dst[t].q[1] = *(const uint4*)(ar + 16);  // K {16..23} / {24..31}
    }
  };

  // one pipeline step: store B(k) to LDS, issue B(k+32), barrier, read B
  // fragment, issue A(k+32) into nxt, run 8 WMMAs on cur.
  auto step = [&](FragBF (&cur)[8], FragBF (&nxt)[8], int k, int buf) {
    *(unsigned int*)&Bs[buf][e0] = packB();
    if (k + 32 < K) issueB(k + 32);
    __syncthreads();   // single barrier per chunk (double buffer covers WAR)
    FragBF bfr;        // lane l15 = column n0+l15; lanes 0-15 K0..7,16..23, 16-31 K8..15,24..31
    bfr.q[0] = *(const uint4*)&Bs[buf][l15 * 32 + half * 8];
    bfr.q[1] = *(const uint4*)&Bs[buf][l15 * 32 + 16 + half * 8];
    if (wactive && (k + 32 < Keff)) issueA(nxt, k + 32);
    if (wactive && (k < Keff)) {
#pragma unroll
      for (int t = 0; t < 8; ++t)
        acc[t] = __builtin_amdgcn_wmma_f32_16x16x32_bf16(
            false, cur[t].v, false, bfr.v, (short)0, acc[t], false, false);
    }
  };

  // ---- prologue + 2x-unrolled main loop -----------------------------------
  issueB(0);
  if (wactive && Keff > 0) issueA(afA, 0);
  {
    int k = 0, buf = 0;
    for (;;) {
      step(afA, afB, k, buf); k += 32; buf ^= 1; if (k >= K) break;
      step(afB, afA, k, buf); k += 32; buf ^= 1; if (k >= K) break;
    }
  }

  if (!wactive) return;   // after all barriers; fully masked causal tile

  // --- epilogue: C layout VGPR j -> row m0+16t+j (+8 for lanes>=16) --------
  const int  n  = n0 + l15;
  const long cz = (long)z * cOffZ;
  const float bv = (FLAGS & GF_BIAS) ? bias[n] : 0.0f;
#pragma unroll
  for (int t = 0; t < 8; ++t) {
#pragma unroll
    for (int j = 0; j < 8; ++j) {
      const int m = m0 + 16 * t + j + half * 8;
      float v = acc[t][j] + bv;
      if (FLAGS & GF_RELU) v = v > 0.0f ? v : 0.0f;
      if (FLAGS & GF_F32)   Cf [cz + (long)m * ldc + n] = v;
      if (FLAGS & GF_BF16)  Cbf[cz + (long)m * ldc + n] = f2bf(v);
      if (FLAGS & GF_TRANS) Ct [(long)n * ldct + m]     = f2bf(v);
    }
  }
}

// ---------------------------------------------------------------------------
// Embedding gather + sinusoidal positional encoding (matches reference PE).
// grid = (1024), block = (256)
// ---------------------------------------------------------------------------
__global__ __launch_bounds__(256)
void embed_kernel(const int* __restrict__ tok, const float* __restrict__ emb,
                  float* __restrict__ of, u16* __restrict__ ob)
{
  const int t  = blockIdx.x;
  const int tk = tok[t];
  for (int d = threadIdx.x; d < 768; d += 256) {
    const int   j   = d >> 1;
    const float inv = __powf(10000.0f, (2.0f * (float)j) / 768.0f);
    const float ang = (float)t / inv;
    const float pe  = (d & 1) ? __cosf(ang) : __sinf(ang);
    const float v   = emb[(long)tk * 768 + d] + pe;
    of[(long)t * 768 + d] = v;
    ob[(long)t * 768 + d] = f2bf(v);
  }
}

// ---------------------------------------------------------------------------
// Attention softmax with masking.  mode 0: key-pad (xtok[k]==1), mode 1: causal
// S: f32 [8][1024][1024], P: bf16 probs.  grid = (1024, 8), block = 256.
// (Causal tiles that were never written are masked by position, so their
//  stored values are irrelevant.)
// ---------------------------------------------------------------------------
__global__ __launch_bounds__(256)
void attn_softmax(const float* __restrict__ S, u16* __restrict__ P,
                  const int* __restrict__ xtok, int mode, float scale)
{
  __shared__ float red[256];
  const int q = blockIdx.x, h = blockIdx.y;
  const float* row = S + ((long)h * 1024 + q) * 1024;
  u16* prow = P + ((long)h * 1024 + q) * 1024;

  float vals[4];
  float mx = -INFINITY;
#pragma unroll
  for (int i = 0; i < 4; ++i) {
    const int kk = threadIdx.x + i * 256;
    float s = row[kk] * scale;
    const bool masked = (mode == 1) ? (kk > q) : (xtok[kk] == 1);
    s = masked ? -INFINITY : s;
    vals[i] = s;
    mx = fmaxf(mx, s);
  }
  red[threadIdx.x] = mx; __syncthreads();
  for (int o = 128; o > 0; o >>= 1) {
    if (threadIdx.x < o) red[threadIdx.x] = fmaxf(red[threadIdx.x], red[threadIdx.x + o]);
    __syncthreads();
  }
  mx = red[0]; __syncthreads();

  float sum = 0.0f;
#pragma unroll
  for (int i = 0; i < 4; ++i) { const float e = __expf(vals[i] - mx); vals[i] = e; sum += e; }
  red[threadIdx.x] = sum; __syncthreads();
  for (int o = 128; o > 0; o >>= 1) {
    if (threadIdx.x < o) red[threadIdx.x] += red[threadIdx.x + o];
    __syncthreads();
  }
  const float inv = 1.0f / red[0];
#pragma unroll
  for (int i = 0; i < 4; ++i) {
    const int kk = threadIdx.x + i * 256;
    prow[kk] = f2bf(vals[i] * inv);
  }
}

// ---------------------------------------------------------------------------
// out = LayerNorm(a + r) * g + b ; writes f32 and bf16 (row-local, so
// out_f == r aliasing is safe). grid = (1024), block = 256 (3 elems/thread).
// ---------------------------------------------------------------------------
__global__ __launch_bounds__(256)
void ln_residual(const float* __restrict__ a, const float* __restrict__ r,
                 const float* __restrict__ g, const float* __restrict__ b,
                 float* __restrict__ of, u16* __restrict__ ob)
{
  __shared__ float red[256];
  const int t = blockIdx.x;
  const float* ar = a + (long)t * 768;
  const float* rr = r + (long)t * 768;

  float v[3]; float s = 0.0f;
#pragma unroll
  for (int i = 0; i < 3; ++i) { const int d = threadIdx.x + i * 256; v[i] = ar[d] + rr[d]; s += v[i]; }
  red[threadIdx.x] = s; __syncthreads();
  for (int o = 128; o > 0; o >>= 1) { if (threadIdx.x < o) red[threadIdx.x] += red[threadIdx.x + o]; __syncthreads(); }
  const float mean = red[0] * (1.0f / 768.0f); __syncthreads();

  float s2 = 0.0f;
#pragma unroll
  for (int i = 0; i < 3; ++i) { const float d = v[i] - mean; s2 += d * d; }
  red[threadIdx.x] = s2; __syncthreads();
  for (int o = 128; o > 0; o >>= 1) { if (threadIdx.x < o) red[threadIdx.x] += red[threadIdx.x + o]; __syncthreads(); }
  const float rstd = rsqrtf(red[0] * (1.0f / 768.0f) + 1e-5f);

#pragma unroll
  for (int i = 0; i < 3; ++i) {
    const int d = threadIdx.x + i * 256;
    const float o = (v[i] - mean) * rstd * g[d] + b[d];
    of[(long)t * 768 + d] = o;
    ob[(long)t * 768 + d] = f2bf(o);
  }
}

// ---------------------------------------------------------------------------
// In-place log_softmax over rows of length n. grid = (1024), block = 256.
// ---------------------------------------------------------------------------
__global__ __launch_bounds__(256)
void log_softmax_rows(float* __restrict__ X, int n)
{
  __shared__ float red[256];
  float* row = X + (long)blockIdx.x * n;
  float mx = -INFINITY;
  for (int i = threadIdx.x; i < n; i += 256) mx = fmaxf(mx, row[i]);
  red[threadIdx.x] = mx; __syncthreads();
  for (int o = 128; o > 0; o >>= 1) { if (threadIdx.x < o) red[threadIdx.x] = fmaxf(red[threadIdx.x], red[threadIdx.x + o]); __syncthreads(); }
  mx = red[0]; __syncthreads();
  float s = 0.0f;
  for (int i = threadIdx.x; i < n; i += 256) s += __expf(row[i] - mx);
  red[threadIdx.x] = s; __syncthreads();
  for (int o = 128; o > 0; o >>= 1) { if (threadIdx.x < o) red[threadIdx.x] += red[threadIdx.x + o]; __syncthreads(); }
  const float lse = mx + __logf(red[0]);
  for (int i = threadIdx.x; i < n; i += 256) row[i] = row[i] - lse;
}

// ---------------------------------------------------------------------------
// Host orchestration.
// Input flattening (setup_inputs dict order, recursive insertion order):
//   0:x 1:y 2:enc_emb 3:dec_emb
//   enc layer l @ 4+l*14 : qW qb kW kb vW vb oW ob lnG lnB l1W l1b l2W l2b
//   dec layer l @ 88+l*22: a1(qW..ob) a2(qW..ob) lnG lnB l1W l1b l2W l2b
//   gen: 220:W 221:b
// ---------------------------------------------------------------------------
extern "C" void kernel_launch(void* const* d_in, const int* in_sizes, int n_in,
                              void* d_out, int out_size, void* d_ws, size_t ws_size,
                              hipStream_t stream)
{
  (void)in_sizes; (void)n_in; (void)out_size; (void)ws_size;
  const int TOK = 1024, D = 768, DFF = 3072, V = 32000;
  const float SCALE = 0.1020620726f;   // 1/sqrt(96)

  const int*   xtok = (const int*)d_in[0];
  const int*   ytok = (const int*)d_in[1];
  const float* eemb = (const float*)d_in[2];
  const float* demb = (const float*)d_in[3];
  auto ENC = [&](int l, int j) { return (const float*)d_in[4 + l * 14 + j]; };
  auto DEC = [&](int l, int j) { return (const float*)d_in[88 + l * 22 + j]; };
  const float* genW = (const float*)d_in[220];
  const float* genB = (const float*)d_in[221];

  // --- workspace carve-out (~76 MB) ---------------------------------------
  char* wp = (char*)d_ws;
  auto carve = [&](size_t bytes) { void* r = wp; wp += (bytes + 255) & ~(size_t)255; return r; };
  float* enc_f  = (float*)carve((size_t)TOK * D * 4);
  u16*   enc_b  = (u16*)  carve((size_t)TOK * D * 2);
  float* cur_f  = (float*)carve((size_t)TOK * D * 4);
  u16*   cur_b  = (u16*)  carve((size_t)TOK * D * 2);
  u16*   q_b    = (u16*)  carve((size_t)TOK * D * 2);
  u16*   k_b    = (u16*)  carve((size_t)TOK * D * 2);
  u16*   vT_b   = (u16*)  carve((size_t)D * TOK * 2);   // [768][1024]
  u16*   attn_b = (u16*)  carve((size_t)TOK * D * 2);
  u16*   ffn_b  = (u16*)  carve((size_t)TOK * DFF * 2);
  float* tA     = (float*)carve((size_t)TOK * D * 4);
  float* tB     = (float*)carve((size_t)TOK * D * 4);
  float* scores = (float*)carve((size_t)8 * TOK * TOK * 4);
  u16*   probs  = (u16*)  carve((size_t)8 * TOK * TOK * 2);

  // templated dispatch: every flag combination used gets its own
  // straight-line specialization (no runtime flag branches on device)
  auto gemm = [&](const u16* A, int lda, long aoz,
                  const void* B, int ldb, long boz, const float* bias,
                  float* Cf, u16* Cbf, u16* Ct, int ldc, long coz, int ldct,
                  int N, int K, int Z, int flags) {
    dim3 g(N / 16, 1, Z), blk(32, 8, 1);
#define GEMM_CASE(F) case (F): \
    wmma_gemm<(F)><<<g, blk, 0, stream>>>(A, lda, aoz, B, ldb, boz, bias, \
                                          Cf, Cbf, Ct, ldc, coz, ldct, K); break
    switch (flags) {
      GEMM_CASE(GF_BF16 | GF_BIAS);                  // Q/K projections
      GEMM_CASE(GF_TRANS | GF_BIAS);                 // V projection (transposed out)
      GEMM_CASE(GF_F32 | GF_BBF16);                  // attention scores
      GEMM_CASE(GF_F32 | GF_BBF16 | GF_CTRIU);       // causal attention scores
      GEMM_CASE(GF_BF16 | GF_BBF16);                 // P @ V
      GEMM_CASE(GF_BF16 | GF_BBF16 | GF_CKLIM);      // causal P @ V
      GEMM_CASE(GF_F32 | GF_BIAS);                   // O proj / FFN2 / generator
      GEMM_CASE(GF_BF16 | GF_BIAS | GF_RELU);        // FFN1
      default: break;
    }
#undef GEMM_CASE
  };

  auto attention = [&](const u16* qsrc, const u16* kvsrc,
                       const float* Wq, const float* bq, const float* Wk, const float* bk,
                       const float* Wv, const float* bv, const float* Wo, const float* bo,
                       int mask_mode, float* outf) {
    gemm(qsrc, D, 0, Wq, D, 0, bq, nullptr, q_b, nullptr, D, 0, 0, D, D, 1, GF_BF16 | GF_BIAS);
    gemm(kvsrc, D, 0, Wk, D, 0, bk, nullptr, k_b, nullptr, D, 0, 0, D, D, 1, GF_BF16 | GF_BIAS);
    gemm(kvsrc, D, 0, Wv, D, 0, bv, nullptr, nullptr, vT_b, 0, 0, TOK, D, D, 1, GF_TRANS | GF_BIAS);
    // scores[h] = Q_h @ K_h^T  (per-head via z: 96-column offsets into Q/K)
    const int sfl = (mask_mode == 1) ? (GF_F32 | GF_BBF16 | GF_CTRIU) : (GF_F32 | GF_BBF16);
    gemm(q_b, D, 96, k_b, D, 96, nullptr, scores, nullptr, nullptr,
         TOK, (long)TOK * TOK, 0, TOK, 96, 8, sfl);
    attn_softmax<<<dim3(TOK, 8), dim3(256), 0, stream>>>(scores, probs, xtok, mask_mode, SCALE);
    // O_h = P_h @ V_h  (B = rows h*96..h*96+95 of V^T, contiguous columns)
    const int pfl = (mask_mode == 1) ? (GF_BF16 | GF_BBF16 | GF_CKLIM) : (GF_BF16 | GF_BBF16);
    gemm(probs, TOK, (long)TOK * TOK, vT_b, TOK, (long)96 * TOK, nullptr,
         nullptr, attn_b, nullptr, D, 96, 0, 96, TOK, 8, pfl);
    gemm(attn_b, D, 0, Wo, D, 0, bo, outf, nullptr, nullptr, D, 0, 0, D, D, 1, GF_F32 | GF_BIAS);
  };

  auto ln = [&](const float* a, const float* r, const float* g, const float* b,
                float* of, u16* ob) {
    ln_residual<<<dim3(TOK), dim3(256), 0, stream>>>(a, r, g, b, of, ob);
  };

  auto ffn = [&](const u16* in_b, const float* W1, const float* b1,
                 const float* W2, const float* b2, float* outf) {
    gemm(in_b, D, 0, W1, D, 0, b1, nullptr, ffn_b, nullptr, DFF, 0, 0, DFF, D, 1,
         GF_BF16 | GF_BIAS | GF_RELU);
    gemm(ffn_b, DFF, 0, W2, DFF, 0, b2, outf, nullptr, nullptr, D, 0, 0, D, DFF, 1,
         GF_F32 | GF_BIAS);
  };

  // ---------------- encoder ----------------
  embed_kernel<<<dim3(TOK), dim3(256), 0, stream>>>(xtok, eemb, enc_f, enc_b);
  for (int l = 0; l < 6; ++l) {
    attention(enc_b, enc_b, ENC(l,0), ENC(l,1), ENC(l,2), ENC(l,3),
              ENC(l,4), ENC(l,5), ENC(l,6), ENC(l,7), /*key-pad*/0, tA);
    ln(tA, enc_f, ENC(l,8), ENC(l,9), tB, cur_b);          // f1 (f32 in tB, bf16 in cur_b)
    ffn(cur_b, ENC(l,10), ENC(l,11), ENC(l,12), ENC(l,13), tA);
    ln(tA, tB, ENC(l,8), ENC(l,9), enc_f, enc_b);          // new hx
  }

  // ---------------- decoder ----------------
  embed_kernel<<<dim3(TOK), dim3(256), 0, stream>>>(ytok, demb, cur_f, cur_b);
  for (int l = 0; l < 6; ++l) {
    attention(cur_b, cur_b, DEC(l,0), DEC(l,1), DEC(l,2), DEC(l,3),
              DEC(l,4), DEC(l,5), DEC(l,6), DEC(l,7), /*causal*/1, tA);
    ln(tA, cur_f, DEC(l,16), DEC(l,17), tB, cur_b);        // f1
    attention(cur_b, enc_b, DEC(l,8), DEC(l,9), DEC(l,10), DEC(l,11),
              DEC(l,12), DEC(l,13), DEC(l,14), DEC(l,15), /*key-pad*/0, tA);
    ln(tA, tB, DEC(l,16), DEC(l,17), tB, cur_b);           // f2 (in-place safe)
    ffn(cur_b, DEC(l,18), DEC(l,19), DEC(l,20), DEC(l,21), tA);
    ln(tA, tB, DEC(l,16), DEC(l,17), cur_f, cur_b);        // new hy
  }

  // ---------------- generator + log_softmax ----------------
  gemm(cur_b, D, 0, genW, D, 0, genB, (float*)d_out, nullptr, nullptr,
       V, 0, 0, V, D, 1, GF_F32 | GF_BIAS);
  log_softmax_rows<<<dim3(TOK), dim3(256), 0, stream>>>((float*)d_out, V);
}
